// CCN_3951369912894
// MI455X (gfx1250) — 16-bit and 8-bit matrix operand order repaired and checked by reading, compiled, not run on hardware
//
#include <hip/hip_runtime.h>

#define NP     4096
#define NREAL  4001
#define DIM    128
#define T2     0.0016f
#define FP8_ONE 0x38

typedef __attribute__((ext_vector_type(16))) _Float16 v16h;
typedef __attribute__((ext_vector_type(8)))  float    v8f;
typedef __attribute__((ext_vector_type(16))) int      v16i;


__global__ void build_locp(const float* __restrict__ nl,
                           const float* __restrict__ depot,
                           float* __restrict__ locp) {
    int i = blockIdx.x * blockDim.x + threadIdx.x;
    if (i >= NP) return;
    float x, y;
    if (i == 0)            { x = depot[0];        y = depot[1];        }
    else if (i < NREAL)    { x = nl[(i-1)*2];     y = nl[(i-1)*2 + 1]; }
    else                   { x = 1.0e9f;          y = 1.0e9f;          }
    locp[i*2]     = x;
    locp[i*2 + 1] = y;
}

__global__ void build_fv0(const float* __restrict__ nl,
                          const float* __restrict__ td,
                          const float* __restrict__ depot,
                          const float* __restrict__ w,
                          const float* __restrict__ b,
                          float* __restrict__ fv0) {
    int idx = blockIdx.x * blockDim.x + threadIdx.x;
    int i = idx / DIM, d = idx % DIM;
    float v = 0.0f;
    if (i < NREAL) {
        float x, y, t;
        if (i == 0) { x = depot[0]; y = depot[1]; t = 0.0f; }
        else        { x = nl[(i-1)*2]; y = nl[(i-1)*2+1]; t = td[i-1]; }
        v = x * w[d*3 + 0] + y * w[d*3 + 1] + t * w[d*3 + 2] + b[d];
        v = v > 0.0f ? v : 0.0f;
    }
    fv0[idx] = v;
}

__global__ void build_A(const float* __restrict__ locp,
                        unsigned char* __restrict__ A8) {
    long long idx = (long long)blockIdx.x * blockDim.x + threadIdx.x;
    int i = (int)(idx >> 12);
    int j = (int)(idx & (NP - 1));
    float dx = locp[i*2]     - locp[j*2];
    float dy = locp[i*2 + 1] - locp[j*2 + 1];
    float d2 = dx*dx + dy*dy;
    A8[idx] = (i < NREAL && j < NREAL && d2 <= T2) ? FP8_ONE : 0;
}

__device__ __forceinline__ v16i load_a_frag(const unsigned char* p) {
    const unsigned int* d = (const unsigned int*)p;
    v16i a;
    a[0]  = (int)d[0];  a[1]  = (int)d[1];  a[2]  = (int)d[4];  a[3]  = (int)d[5];
    a[4]  = (int)d[8];  a[5]  = (int)d[9];  a[6]  = (int)d[12]; a[7]  = (int)d[13];
    a[8]  = (int)d[16]; a[9]  = (int)d[17]; a[10] = (int)d[20]; a[11] = (int)d[21];
    a[12] = (int)d[24]; a[13] = (int)d[25]; a[14] = (int)d[28]; a[15] = (int)d[29];
    return a;
}
__device__ __forceinline__ v16i load_b_frag(const unsigned char* p) {
    const unsigned int* d = (const unsigned int*)p;
    v16i b;
    b[0]  = (int)d[0];  b[1]  = (int)d[1];  b[2]  = (int)d[2];  b[3]  = (int)d[3];
    b[4]  = (int)d[8];  b[5]  = (int)d[9];  b[6]  = (int)d[10]; b[7]  = (int)d[11];
    b[8]  = (int)d[16]; b[9]  = (int)d[17]; b[10] = (int)d[18]; b[11] = (int)d[19];
    b[12] = (int)d[24]; b[13] = (int)d[25]; b[14] = (int)d[26]; b[15] = (int)d[27];
    return b;
}

template<int MODE>
__global__ void gemm_fp8(const unsigned char* __restrict__ X,
                         const unsigned char* __restrict__ Bsym,
                         unsigned char* __restrict__ M2out,
                         float* __restrict__ Wout,
                         const unsigned char* __restrict__ M2in) {
    int wave = (blockIdx.x * blockDim.x + threadIdx.x) >> 5;
    int lane = threadIdx.x & 31;
    int ti = wave >> 7;
    int tj = wave & 127;
    int lm = lane & 15;
    bool hi = lane >= 16;

    const unsigned char* arow0 = X    + (size_t)(ti*32 + lm) * NP + (hi ? 8 : 0);
    const unsigned char* arow1 = arow0 + (size_t)16 * NP;
    const unsigned char* brow0 = Bsym + (size_t)(tj*32 + lm) * NP + (hi ? 8 : 0);
    const unsigned char* brow1 = brow0 + (size_t)16 * NP;

    v8f c00 = {}, c01 = {}, c10 = {}, c11 = {};
    for (int kb = 0; kb < NP; kb += 128) {
        __builtin_prefetch(arow0 + kb + 512, 0, 1);
        __builtin_prefetch(brow0 + kb + 512, 0, 1);
        v16i a0 = load_a_frag(arow0 + kb);
        v16i a1 = load_a_frag(arow1 + kb);
        v16i b0 = load_a_frag(brow0 + kb);
        v16i b1 = load_a_frag(brow1 + kb);
        c00 = __builtin_amdgcn_wmma_f32_16x16x128_fp8_fp8(a0, b0, (short)0, c00, false, false);
        c01 = __builtin_amdgcn_wmma_f32_16x16x128_fp8_fp8(a0, b1, (short)0, c01, false, false);
        c10 = __builtin_amdgcn_wmma_f32_16x16x128_fp8_fp8(a1, b0, (short)0, c10, false, false);
        c11 = __builtin_amdgcn_wmma_f32_16x16x128_fp8_fp8(a1, b1, (short)0, c11, false, false);
    }

    int col0 = tj*32 + lm;
    int col1 = col0 + 16;
    #pragma unroll
    for (int r = 0; r < 8; ++r) {
        int row0 = ti*32 + r + (hi ? 8 : 0);
        int row1 = row0 + 16;
        size_t o00 = (size_t)row0 * NP + col0, o01 = (size_t)row0 * NP + col1;
        size_t o10 = (size_t)row1 * NP + col0, o11 = (size_t)row1 * NP + col1;
        if (MODE == 0) {
            M2out[o00] = (c00[r] > 0.5f) ? FP8_ONE : 0;
            M2out[o01] = (c01[r] > 0.5f) ? FP8_ONE : 0;
            M2out[o10] = (c10[r] > 0.5f) ? FP8_ONE : 0;
            M2out[o11] = (c11[r] > 0.5f) ? FP8_ONE : 0;
        } else {
            Wout[o00] = M2in[o00] ? c00[r] : 0.0f;
            Wout[o01] = M2in[o01] ? c01[r] : 0.0f;
            Wout[o10] = M2in[o10] ? c10[r] : 0.0f;
            Wout[o11] = M2in[o11] ? c11[r] : 0.0f;
        }
    }
}

template<bool LEFT_I8>
__global__ void gemm_f16(const void* __restrict__ Lv,
                         const float* __restrict__ R,
                         float* __restrict__ Out) {
    int wave = (blockIdx.x * blockDim.x + threadIdx.x) >> 5;
    int lane = threadIdx.x & 31;
    int ti = wave >> 3;
    int tc = wave & 7;
    int lm = lane & 15;
    bool hi = lane >= 16;
    int s  = hi ? 8 : 0;
    int bs = hi ? 16 : 0;

    const unsigned char* L8 = (const unsigned char*)Lv;
    const float*         Lf = (const float*)Lv;
    size_t lrow = (size_t)(ti*16 + lm) * NP;

    v8f acc = {};
    for (int kb = 0; kb < NP; kb += 32) {
        v16h a, b;
        #pragma unroll
        for (int e = 0; e < 8; ++e) {
            int k0 = kb + s + e;
            int k1 = kb + 16 + s + e;
            float f0, f1;
            if (LEFT_I8) { f0 = L8[lrow + k0] ? 1.0f : 0.0f;
                           f1 = L8[lrow + k1] ? 1.0f : 0.0f; }
            else         { f0 = Lf[lrow + k0];  f1 = Lf[lrow + k1]; }
            a[e]     = (_Float16)f0;
            a[e + 8] = (_Float16)f1;
        }
        #pragma unroll
        for (int e = 0; e < 16; ++e) {
            int k = kb + s + e + (e >= 8 ? 8 : 0);
            b[e] = (_Float16)R[(size_t)k * DIM + tc*16 + lm];
        }
        acc = __builtin_amdgcn_wmma_f32_16x16x32_f16(false, a, false, b,
                                                     (short)0, acc, false, false);
    }

    int col = tc*16 + lm;
    #pragma unroll
    for (int r = 0; r < 8; ++r) {
        int row = ti*16 + r + (hi ? 8 : 0);
        Out[(size_t)row * DIM + col] = acc[r];
    }
}

__global__ void copy_out(const float* __restrict__ fv2,
                         float* __restrict__ out, int n) {
    int i = blockIdx.x * blockDim.x + threadIdx.x;
    if (i < n) out[i] = fv2[i];
}

extern "C" void kernel_launch(void* const* d_in, const int* in_sizes, int n_in,
                              void* d_out, int out_size, void* d_ws, size_t ws_size,
                              hipStream_t stream) {
    const float* nl    = (const float*)d_in[0];
    const float* td    = (const float*)d_in[1];
    const float* depot = (const float*)d_in[2];
    const float* w0w   = (const float*)d_in[3];
    const float* w0b   = (const float*)d_in[4];

    char* ws = (char*)d_ws;
    unsigned char* A8   = (unsigned char*)(ws);
    unsigned char* M28  = (unsigned char*)(ws + (size_t)16777216);
    float*         W32  = (float*)(ws + (size_t)33554432);
    float*         fv0  = (float*)(ws + (size_t)100663296);
    float*         fv1  = (float*)(ws + (size_t)100663296 + 2097152);
    float*         fv2  = (float*)(ws + (size_t)100663296 + 2*2097152);
    float*         locp = (float*)(ws + (size_t)100663296 + 3*2097152);

    build_locp<<<NP/256, 256, 0, stream>>>(nl, depot, locp);
    build_fv0<<<(NP*DIM)/256, 256, 0, stream>>>(nl, td, depot, w0w, w0b, fv0);

    build_A<<<(NP*NP)/256, 256, 0, stream>>>(locp, A8);

    gemm_fp8<0><<<2048, 256, 0, stream>>>(A8, A8, M28, nullptr, nullptr);

    gemm_f16<true><<<256, 256, 0, stream>>>(A8, fv0, fv1);

    gemm_fp8<1><<<2048, 256, 0, stream>>>(M28, A8, nullptr, W32, M28);

    gemm_f16<false><<<256, 256, 0, stream>>>(W32, fv1, fv2);

    copy_out<<<(out_size + 255)/256, 256, 0, stream>>>(fv2, (float*)d_out, out_size);
}
